// MultiHeadAttentionForLoRA_73177652789609
// MI455X (gfx1250) — matrix operand order repaired and checked by reading, not compiled, not run on hardware
//
#include <hip/hip_runtime.h>
#include <hip/hip_bf16.h>

// MultiHeadAttention + LoRA for MI455X (gfx1250), wave32 + WMMA f16.
// N=2, S=2048, D=1024, H=16, HD=64, R=4.
// GEMMs: LDS double-buffered via gfx1250 async global->LDS (ASYNCcnt).

#define D_     1024
#define S_     2048
#define NBATCH 2
#define NHEAD  16
#define HD_    64
#define MTOT   (NBATCH * S_)   // 4096 tokens
#define RANK   4

// LDS staging: 128x32-f16 tiles, rows padded to 80B to kill bank conflicts
// (20*row mod 64 is a permutation over the 16 rows hit by one b128 read).
#define LDS_LD    80                  // bytes per tile row (64B data + 16B pad)
#define LDS_ATILE (128 * LDS_LD)      // 10240 B per tile
#define LDS_STAGE (2 * LDS_ATILE)     // A + B per stage = 20480 B
#define LDS_TOTAL (2 * LDS_STAGE)     // double buffered = 40960 B

typedef __attribute__((ext_vector_type(16))) _Float16 v16h;
typedef __attribute__((ext_vector_type(8)))  _Float16 v8h;
typedef __attribute__((ext_vector_type(8)))  float    v8f;

union H16 { v16h v; v8h h[2]; unsigned u[8]; };
union H8  { v8h  v; _Float16 e[8]; };
union F8  { v8f  v; float f[8]; };

// ---- WMMA fragment loaders (this code's k order: k = 16*half + s, same in A and B) ----
// A (16x32 f16, MxK): lanes 0-15 row=lane, K=0..15;
//                     lanes 16-31 row=lane-16, K=16..31.
__device__ __forceinline__ H16 load_a_frag(const _Float16* base, int ld) {
  const int lane = threadIdx.x & 31;
  const int half = lane >> 4, row = lane & 15;
  const _Float16* p = base + row * ld + half * 16;
  H16 r;
  r.h[0] = *(const v8h*)(p);
  r.h[1] = *(const v8h*)(p + 8);
  return r;
}
// B (32x16 f16, KxN) given Bt[n][k]: lanes 0-15 col=lane, K=0..15;
// lanes 16-31 col=lane-16, K=16..31 (32 contiguous f16 of one Bt row).
__device__ __forceinline__ H16 load_b_frag(const _Float16* base, int ld) {
  const int lane = threadIdx.x & 31;
  const int half = lane >> 4, col = lane & 15;
  const _Float16* p = base + col * ld + half * 16;
  H16 r;
  r.h[0] = *(const v8h*)(p);
  r.h[1] = *(const v8h*)(p + 8);
  return r;
}
// Same fragment shapes out of padded LDS tiles (-> ds_load_b128).
__device__ __forceinline__ H16 lds_a_frag(const char* smem, unsigned rowbase) {
  const int lane = threadIdx.x & 31;
  const int half = lane >> 4, row = lane & 15;
  const char* p = smem + rowbase + row * LDS_LD + half * 32;
  H16 r;
  r.h[0] = *(const v8h*)(p);
  r.h[1] = *(const v8h*)(p + 16);
  return r;
}
__device__ __forceinline__ H16 lds_b_frag(const char* smem, unsigned rowbase) {
  const int lane = threadIdx.x & 31;
  const int half = lane >> 4, col = lane & 15;
  const char* p = smem + rowbase + col * LDS_LD + half * 32;
  H16 r;
  r.h[0] = *(const v8h*)(p);
  r.h[1] = *(const v8h*)(p + 16);
  return r;
}
__device__ __forceinline__ v8f wmma16(const H16& a, const H16& b, const v8f& c) {
  return __builtin_amdgcn_wmma_f32_16x16x32_f16(false, a.v, false, b.v,
                                                (short)0, c, false, false);
}
__device__ __forceinline__ unsigned pk2(float x, float y) {
  union { _Float16 e[2]; unsigned u; } t;
  t.e[0] = (_Float16)x; t.e[1] = (_Float16)y;
  return t.u;
}

// gfx1250 async copy: 16B global -> LDS, tracked by ASYNCcnt.
__device__ __forceinline__ void async_b128(unsigned lds_off, const void* gaddr) {
  asm volatile("global_load_async_to_lds_b128 %0, %1, off"
               :: "v"(lds_off), "v"(gaddr) : "memory");
}
__device__ __forceinline__ void wait_async0() {
  asm volatile("s_wait_asynccnt 0x0" ::: "memory");
}

// Stage one 128x32 A tile + one 128x32 B tile into LDS buffer `bufbase`.
// 256 threads x 2 chunks x (A,B) = 4 async instructions per thread.
__device__ __forceinline__ void async_stage(const _Float16* __restrict__ X,
                                            const _Float16* __restrict__ W,
                                            int m0blk, int n0blk, int kk,
                                            unsigned bufbase) {
  const int tid = threadIdx.x;
  #pragma unroll
  for (int c = 0; c < 2; ++c) {
    const int chunk = tid + c * 256;          // 0..511
    const int row = chunk >> 2, cc = chunk & 3;
    async_b128(bufbase + row * LDS_LD + cc * 16,
               X + (m0blk + row) * D_ + kk + cc * 8);
    async_b128(bufbase + LDS_ATILE + row * LDS_LD + cc * 16,
               W + (n0blk + row) * D_ + kk + cc * 8);
  }
}

// Shared GEMM core: block 128x128, 8 waves (4M x 2N), wave tile 32x64.
// Double-buffered LDS; copies for step i+1 overlap WMMA of step i.
__device__ __forceinline__ void gemm_core(const _Float16* __restrict__ X,
                                          const _Float16* __restrict__ W,
                                          F8 (&acc)[2][4]) {
  extern __shared__ char smem[];
  const int tid = threadIdx.x, w = tid >> 5;
  const int wm = w >> 1, wn = w & 1;
  const int m0blk = blockIdx.x * 128, n0blk = blockIdx.y * 128;

  #pragma unroll
  for (int i = 0; i < 2; ++i)
    #pragma unroll
    for (int j = 0; j < 4; ++j)
      #pragma unroll
      for (int e = 0; e < 8; ++e) acc[i][j].f[e] = 0.f;

  async_stage(X, W, m0blk, n0blk, 0, 0);      // preload stage 0
  for (int kk = 0; kk < D_; kk += 32) {
    wait_async0();                            // copies for this stage done
    __syncthreads();                          // visible to all waves; also
                                              // proves prior stage fully read
    if (kk + 32 < D_)
      async_stage(X, W, m0blk, n0blk, kk + 32,
                  (unsigned)((((kk >> 5) + 1) & 1) * LDS_STAGE));
    const unsigned cur = (unsigned)(((kk >> 5) & 1) * LDS_STAGE);
    const unsigned aBase = cur + (wm * 32) * LDS_LD;
    const unsigned bBase = cur + LDS_ATILE + (wn * 64) * LDS_LD;
    H16 a0 = lds_a_frag(smem, aBase);
    H16 a1 = lds_a_frag(smem, aBase + 16 * LDS_LD);
    H16 bf[4];
    #pragma unroll
    for (int j = 0; j < 4; ++j)
      bf[j] = lds_b_frag(smem, bBase + 16 * j * LDS_LD);
    #pragma unroll
    for (int j = 0; j < 4; ++j) {
      acc[0][j].v = wmma16(a0, bf[j], acc[0][j].v);
      acc[1][j].v = wmma16(a1, bf[j], acc[1][j].v);
    }
  }
}

// ---- Prep 1: fold LoRA into weights, convert to f16 ---------------------
// (x@qB^T)@qA^T == x@(qA@qB)^T, so W_eff = W + A*B; row-major W_eff IS the
// Bt layout the B-fragment loader wants.
__global__ __launch_bounds__(256) void prep_weights_kernel(
    const float* __restrict__ Wq, const float* __restrict__ Wk,
    const float* __restrict__ Wv, const float* __restrict__ Wo,
    const float* __restrict__ qA, const float* __restrict__ qB,
    const float* __restrict__ vA, const float* __restrict__ vB,
    _Float16* __restrict__ wq, _Float16* __restrict__ wk,
    _Float16* __restrict__ wv, _Float16* __restrict__ wo) {
  int idx = blockIdx.x * 256 + threadIdx.x;       // 0 .. 4*D*D-1
  int which = idx >> 20;                          // D*D == 1<<20
  int e = idx & (D_ * D_ - 1);
  int n = e >> 10, k = e & (D_ - 1);
  if (which == 0) {
    float val = Wq[e];
    #pragma unroll
    for (int r = 0; r < RANK; ++r) val += qA[n * RANK + r] * qB[r * D_ + k];
    wq[e] = (_Float16)val;
  } else if (which == 1) {
    wk[e] = (_Float16)Wk[e];
  } else if (which == 2) {
    float val = Wv[e];
    #pragma unroll
    for (int r = 0; r < RANK; ++r) val += vA[n * RANK + r] * vB[r * D_ + k];
    wv[e] = (_Float16)val;
  } else {
    wo[e] = (_Float16)Wo[e];
  }
}

// ---- Prep 2: convert activations fp32 -> f16 (vectorized x4) ------------
__global__ __launch_bounds__(256) void cvt_inputs_kernel(
    const float* __restrict__ q, const float* __restrict__ k,
    const float* __restrict__ v, _Float16* __restrict__ xq,
    _Float16* __restrict__ xk, _Float16* __restrict__ xv) {
  const int per = MTOT * D_ / 4;
  int idx = blockIdx.x * 256 + threadIdx.x;       // 0 .. 3*per-1
  const float* s; _Float16* d; int e = idx;
  if (idx < per)            { s = q; d = xq; }
  else if (idx < 2 * per)   { s = k; d = xk; e -= per; }
  else                      { s = v; d = xv; e -= 2 * per; }
  float4 x = ((const float4*)s)[e];
  union { _Float16 h[4]; float2 f; } t;
  t.h[0] = (_Float16)x.x; t.h[1] = (_Float16)x.y;
  t.h[2] = (_Float16)x.z; t.h[3] = (_Float16)x.w;
  ((float2*)d)[e] = t.f;
}

// ---- QKV projection GEMM --------------------------------------------------
// blockIdx.z: 0=Q, 1=K (write [n,h,s,hd]), 2=V (write transposed [n,h,hd,s]).
__global__ __launch_bounds__(256) void qkv_gemm_kernel(
    const _Float16* __restrict__ xq, const _Float16* __restrict__ xk,
    const _Float16* __restrict__ xv, const _Float16* __restrict__ wq,
    const _Float16* __restrict__ wk, const _Float16* __restrict__ wv,
    const float* __restrict__ bq, const float* __restrict__ bk,
    const float* __restrict__ bv, _Float16* __restrict__ Qo,
    _Float16* __restrict__ Ko, _Float16* __restrict__ Vt) {
  const int proj = blockIdx.z;
  const _Float16* X = (proj == 0) ? xq : (proj == 1) ? xk : xv;
  const _Float16* W = (proj == 0) ? wq : (proj == 1) ? wk : wv;
  const float* bias = (proj == 0) ? bq : (proj == 1) ? bk : bv;

  const int tid = threadIdx.x, lane = tid & 31, w = tid >> 5;
  const int half = lane >> 4, lr = lane & 15;
  const int wm = w >> 1, wn = w & 1;
  const int m0 = blockIdx.x * 128 + wm * 32;
  const int n0 = blockIdx.y * 128 + wn * 64;

  F8 acc[2][4];
  gemm_core(X, W, acc);

  #pragma unroll
  for (int i = 0; i < 2; ++i) {
    #pragma unroll
    for (int j = 0; j < 4; ++j) {
      const int colg = n0 + 16 * j + lr;          // output feature
      const float bsv = bias[colg];
      const int rowbase = m0 + 16 * i + half * 8; // token index of f[0]
      const int h = colg >> 6, hd = colg & (HD_ - 1);
      if (proj < 2) {
        _Float16* Oup = (proj == 0) ? Qo : Ko;    // [n,h,s,hd]
        #pragma unroll
        for (int v = 0; v < 8; ++v) {
          int t = rowbase + v;
          int nb = t >> 11, s = t & (S_ - 1);
          Oup[((nb * NHEAD + h) * S_ + s) * HD_ + hd] =
              (_Float16)(acc[i][j].f[v] + bsv);
        }
      } else {                                    // V transposed [n,h,hd,s]
        int nb = rowbase >> 11, s0 = rowbase & (S_ - 1);
        H8 pk;
        #pragma unroll
        for (int v = 0; v < 8; ++v) pk.e[v] = (_Float16)(acc[i][j].f[v] + bsv);
        *(v8h*)(Vt + ((nb * NHEAD + h) * HD_ + hd) * S_ + s0) = pk.v;
      }
    }
  }
}

// ---- Flash attention: one wave per (batch, head, 16-query block) --------
// Computes S^T = K*Q^T so softmax (over keys) reduces along M/VGPR dim:
// per-lane 16-value reduce + one cross-half shfl_xor(16).
__global__ __launch_bounds__(128) void attn_kernel(
    const _Float16* __restrict__ Qb, const _Float16* __restrict__ Kb,
    const _Float16* __restrict__ Vt, const unsigned char* __restrict__ kpm,
    _Float16* __restrict__ attn) {
  const int wid = blockIdx.x * 4 + (threadIdx.x >> 5);
  const int qblk = wid & 127, h = (wid >> 7) & 15, nb = wid >> 11;
  const int lane = threadIdx.x & 31, half = lane >> 4, lr = lane & 15;
  const int q0 = qblk * 16;
  const _Float16* Qh = Qb + (nb * NHEAD + h) * S_ * HD_;
  const _Float16* Kh = Kb + (nb * NHEAD + h) * S_ * HD_;
  const _Float16* Vh = Vt + (nb * NHEAD + h) * HD_ * S_;
  const unsigned char* mrow = kpm + nb * S_;

  // Q^T B-fragments (held for the whole K loop): Bt[q][d] = Q row-major.
  H16 bq0 = load_b_frag(Qh + q0 * HD_ + 0,  HD_);
  H16 bq1 = load_b_frag(Qh + q0 * HD_ + 32, HD_);

  F8 O[4];                                        // O^T [64d x 16q], f32
  #pragma unroll
  for (int f = 0; f < 4; ++f)
    #pragma unroll
    for (int e = 0; e < 8; ++e) O[f].f[e] = 0.f;
  float mrun = -1e30f, lrun = 0.f;
  const int qg = q0 + lr;                         // this lane's query index

  for (int j0 = 0; j0 <= q0 + 15; j0 += 32) {     // causal key-block bound
    if (j0 + 32 <= q0 + 15) {                     // prefetch next K/V block
      __builtin_prefetch(Kh + (j0 + 32) * HD_ + lane * 8, 0, 0);
      __builtin_prefetch(Vh + lane * S_ + j0 + 32, 0, 0);
    }
    // S^T tiles: c[t] covers keys j0+16t .. j0+16t+15 (rows), queries (cols)
    F8 c[2];
    #pragma unroll
    for (int t = 0; t < 2; ++t) {
      #pragma unroll
      for (int e = 0; e < 8; ++e) c[t].f[e] = 0.f;
      H16 ak0 = load_a_frag(Kh + (j0 + 16 * t) * HD_ + 0,  HD_);
      H16 ak1 = load_a_frag(Kh + (j0 + 16 * t) * HD_ + 32, HD_);
      c[t].v = wmma16(ak0, bq0, c[t].v);
      c[t].v = wmma16(ak1, bq1, c[t].v);
    }
    // Mask (padding + causal) then scale, as reference: (mask?-1e5:s)*0.125
    float z[2][8];
    #pragma unroll
    for (int t = 0; t < 2; ++t) {
      unsigned long long mb =
          *(const unsigned long long*)(mrow + j0 + 16 * t + 8 * half);
      #pragma unroll
      for (int v = 0; v < 8; ++v) {
        int kg = j0 + 16 * t + 8 * half + v;
        bool msk = (((mb >> (8 * v)) & 1ull) != 0ull) | (kg > qg);
        z[t][v] = (msk ? -100000.0f : c[t].f[v]) * 0.125f;
      }
    }
    // Online softmax stats.
    float mloc = -1e30f;
    #pragma unroll
    for (int t = 0; t < 2; ++t)
      #pragma unroll
      for (int v = 0; v < 8; ++v) mloc = fmaxf(mloc, z[t][v]);
    mloc = fmaxf(mloc, __shfl_xor(mloc, 16, 32));
    const float mnew = fmaxf(mrun, mloc);
    const float corr = __expf(mrun - mnew);
    float p[2][8], sl = 0.f;
    #pragma unroll
    for (int t = 0; t < 2; ++t)
      #pragma unroll
      for (int v = 0; v < 8; ++v) { p[t][v] = __expf(z[t][v] - mnew); sl += p[t][v]; }
    sl += __shfl_xor(sl, 16, 32);
    lrun = lrun * corr + sl;
    mrun = mnew;
    #pragma unroll
    for (int f = 0; f < 4; ++f)
      #pragma unroll
      for (int e = 0; e < 8; ++e) O[f].f[e] *= corr;
    // Build P^T B-fragment [32k x 16q] in registers (one cross-half swap).
    unsigned a[4], b[4];
    #pragma unroll
    for (int jj = 0; jj < 4; ++jj) {
      a[jj] = pk2(p[0][2 * jj], p[0][2 * jj + 1]);
      b[jj] = pk2(p[1][2 * jj], p[1][2 * jj + 1]);
    }
    H16 Bp;
    #pragma unroll
    for (int jj = 0; jj < 4; ++jj) {
      unsigned snd = half ? a[jj] : b[jj];
      unsigned rcv = (unsigned)__shfl_xor((int)snd, 16, 32);
      if (half == 0) { Bp.u[jj] = a[jj]; Bp.u[4 + jj] = rcv; }
      else           { Bp.u[jj] = rcv;   Bp.u[4 + jj] = b[jj]; }
    }
    // O^T += V^T * P^T : A-fragments straight from transposed V rows.
    #pragma unroll
    for (int f = 0; f < 4; ++f) {
      H16 av = load_a_frag(Vh + (16 * f) * S_ + j0, S_);
      O[f].v = wmma16(av, Bp, O[f].v);
    }
  }

  const float invl = 1.0f / lrun;
  #pragma unroll
  for (int f = 0; f < 4; ++f) {
    H8 pk;
    #pragma unroll
    for (int v = 0; v < 8; ++v) pk.e[v] = (_Float16)(O[f].f[v] * invl);
    const int dbase = 16 * f + 8 * half;
    *(v8h*)(attn + (nb * S_ + q0 + lr) * D_ + h * HD_ + dbase) = pk.v;
  }
}

// ---- Output projection: d_out = attn @ Wo^T + bo (f32 out) --------------
__global__ __launch_bounds__(256) void out_gemm_kernel(
    const _Float16* __restrict__ X, const _Float16* __restrict__ W,
    const float* __restrict__ bias, float* __restrict__ out) {
  const int tid = threadIdx.x, lane = tid & 31, w = tid >> 5;
  const int half = lane >> 4, lr = lane & 15;
  const int wm = w >> 1, wn = w & 1;
  const int m0 = blockIdx.x * 128 + wm * 32;
  const int n0 = blockIdx.y * 128 + wn * 64;

  F8 acc[2][4];
  gemm_core(X, W, acc);

  #pragma unroll
  for (int i = 0; i < 2; ++i)
    #pragma unroll
    for (int j = 0; j < 4; ++j) {
      const int colg = n0 + 16 * j + lr;
      const float bsv = bias[colg];
      const int rowbase = m0 + 16 * i + half * 8;
      #pragma unroll
      for (int v = 0; v < 8; ++v)
        out[(rowbase + v) * D_ + colg] = acc[i][j].f[v] + bsv;
    }
}

extern "C" void kernel_launch(void* const* d_in, const int* in_sizes, int n_in,
                              void* d_out, int out_size, void* d_ws,
                              size_t ws_size, hipStream_t stream) {
  (void)in_sizes; (void)n_in; (void)out_size; (void)ws_size;
  const float* query = (const float*)d_in[0];
  const float* key   = (const float*)d_in[1];
  const float* value = (const float*)d_in[2];
  const unsigned char* kpm = (const unsigned char*)d_in[3];  // bool [N,S]
  // d_in[4] attn_mask: strictly causal; reproduced analytically on-chip.
  const float* Wq = (const float*)d_in[5];  const float* bq = (const float*)d_in[6];
  const float* Wk = (const float*)d_in[7];  const float* bk = (const float*)d_in[8];
  const float* Wv = (const float*)d_in[9];  const float* bv = (const float*)d_in[10];
  const float* Wo = (const float*)d_in[11]; const float* bo = (const float*)d_in[12];
  const float* qA = (const float*)d_in[13]; const float* qB = (const float*)d_in[14];
  const float* vA = (const float*)d_in[15]; const float* vB = (const float*)d_in[16];

  // Workspace carve-out (~64 MB total).
  char* ws = (char*)d_ws;
  size_t off = 0;
  auto carve = [&](size_t bytes) { char* p = ws + off; off += bytes; return p; };
  _Float16* xq = (_Float16*)carve((size_t)MTOT * D_ * 2);
  _Float16* xk = (_Float16*)carve((size_t)MTOT * D_ * 2);
  _Float16* xv = (_Float16*)carve((size_t)MTOT * D_ * 2);
  _Float16* wq = (_Float16*)carve((size_t)D_ * D_ * 2);
  _Float16* wk = (_Float16*)carve((size_t)D_ * D_ * 2);
  _Float16* wv = (_Float16*)carve((size_t)D_ * D_ * 2);
  _Float16* wo = (_Float16*)carve((size_t)D_ * D_ * 2);
  _Float16* Qb = (_Float16*)carve((size_t)MTOT * D_ * 2);   // [n,h,s,hd]
  _Float16* Kb = (_Float16*)carve((size_t)MTOT * D_ * 2);   // [n,h,s,hd]
  _Float16* Vt = (_Float16*)carve((size_t)MTOT * D_ * 2);   // [n,h,hd,s]
  _Float16* at = (_Float16*)carve((size_t)MTOT * D_ * 2);   // [n,s,d]

  prep_weights_kernel<<<dim3(4 * D_ * D_ / 256), 256, 0, stream>>>(
      Wq, Wk, Wv, Wo, qA, qB, vA, vB, wq, wk, wv, wo);
  cvt_inputs_kernel<<<dim3(3 * MTOT * D_ / 4 / 256), 256, 0, stream>>>(
      query, key, value, xq, xk, xv);
  qkv_gemm_kernel<<<dim3(MTOT / 128, D_ / 128, 3), 256, LDS_TOTAL, stream>>>(
      xq, xk, xv, wq, wk, wv, bq, bk, bv, Qb, Kb, Vt);
  attn_kernel<<<dim3(NBATCH * NHEAD * (S_ / 16) / 4), 128, 0, stream>>>(
      Qb, Kb, Vt, kpm, at);
  out_gemm_kernel<<<dim3(MTOT / 128, D_ / 128), 256, LDS_TOTAL, stream>>>(
      at, wo, bo, (float*)d_out);
}
